// CausalMultiheadSelfAttention_71760313581991
// MI455X (gfx1250) — compile-verified, matrix operand order checked by reading, not run on hardware
//
#include <hip/hip_runtime.h>

// ---------------------------------------------------------------------------
// CDNA5 (gfx1250) causal multi-head self-attention, f16 WMMA path, f32 accum.
// B=4, S=2048, D=1024, H=16, DK=64
//
// Roofline: ~103 GFLOP total vs ~100-150 MB of HBM traffic (fits in 192MB L2)
// => compute bound => everything runs on v_wmma_f32_16x16x32_f16.
// Attention tiles are staged into LDS with the gfx1250 async-to-LDS engine
// (ASYNCcnt), double-buffered one key-block ahead of the WMMAs.
// ---------------------------------------------------------------------------

typedef _Float16 v16h __attribute__((ext_vector_type(16)));
typedef _Float16 v8h  __attribute__((ext_vector_type(8)));
typedef float    v8f  __attribute__((ext_vector_type(8)));

union U16h { v16h v; _Float16 e[16]; };
union U8f  { v8f  v; float    e[8];  };

__device__ __forceinline__ v8f wmma_f16(v16h a, v16h b, v8f c) {
  // D = A(16x32 f16) * B(32x16 f16) + C(16x16 f32)
  return __builtin_amdgcn_wmma_f32_16x16x32_f16(
      /*neg_a=*/false, a, /*neg_b=*/false, b,
      /*c_mod=*/(short)0, c, /*reuse_a=*/false, /*reuse_b=*/false);
}

// Per-lane operand fetch for the 16-bit A/B WMMA layout: chunk0 = K[0..7],
// chunk1 = K[16..23] relative to p (lane row and koff folded in by caller).
__device__ __forceinline__ v16h ld_ab(const _Float16* p) {
  v8h lo = *(const v8h*)(p);
  v8h hi = *(const v8h*)(p + 16);
  return __builtin_shufflevector(lo, hi, 0, 1, 2, 3, 4, 5, 6, 7,
                                 8, 9, 10, 11, 12, 13, 14, 15);
}

// ---- gfx1250 async global->LDS copy (ASYNCcnt) ----------------------------
__device__ __forceinline__ void async_copy_b128(unsigned lds_off,
                                                const _Float16* gaddr) {
  asm volatile("global_load_async_to_lds_b128 %0, %1, off"
               :: "v"(lds_off), "v"(gaddr) : "memory");
}
__device__ __forceinline__ void wait_async_0() {
  asm volatile("s_wait_asynccnt 0x0" ::: "memory");
}
__device__ __forceinline__ void wait_async_16() {
  asm volatile("s_wait_asynccnt 0x10" ::: "memory");
}

// ---------------------------------------------------------------------------
// f32 -> f16 conversion
// ---------------------------------------------------------------------------
__global__ void __launch_bounds__(256) f32_to_f16(const float* __restrict__ in,
                                                  _Float16* __restrict__ out,
                                                  int n) {
  int i = blockIdx.x * blockDim.x + threadIdx.x;
  if (i < n) out[i] = (_Float16)in[i];
}

// ---------------------------------------------------------------------------
// GEMM: C[M][N] = A[M][K] * Bw[N][K]^T   (i.e. x @ w.T)
// One wave computes a 16(M) x 64(N) tile; A regs reused across 4 N-tiles.
// Output modes: f32 (Cf != null), f16 (Ch, vtrans==0), or f16 stored
// per-head-transposed Vt[b][h][dk][s] (Ch, vtrans==1; dims hardcoded).
// ---------------------------------------------------------------------------
__global__ void __launch_bounds__(256) gemm_wmma_f16(
    const _Float16* __restrict__ A, const _Float16* __restrict__ Bw,
    float* __restrict__ Cf, _Float16* __restrict__ Ch, int vtrans,
    int M, int N, int K) {
  const int lane = threadIdx.x & 31;
  const int wid  = blockIdx.x * (blockDim.x >> 5) + (threadIdx.x >> 5);
  const int mtiles = M >> 4;
  const int mt = wid % mtiles;
  const int nt = wid / mtiles;
  if (nt >= (N >> 6)) return;

  const int row  = lane & 15;        // M-row (A) / N-row (B) within tile
  const int hf   = lane >> 4;
  const int koff = hf << 3;          // K sub-chunk select per wave32 layout

  const _Float16* arow = A  + (size_t)(mt * 16 + row) * K + koff;
  const _Float16* brow = Bw + (size_t)(nt * 64 + row) * K + koff;
  const size_t bstride = (size_t)16 * K;

  v8f acc0 = {}, acc1 = {}, acc2 = {}, acc3 = {};
  for (int k = 0; k < K; k += 32) {
    const v16h a = ld_ab(arow + k);
    acc0 = wmma_f16(a, ld_ab(brow + k),               acc0);
    acc1 = wmma_f16(a, ld_ab(brow + bstride + k),     acc1);
    acc2 = wmma_f16(a, ld_ab(brow + 2 * bstride + k), acc2);
    acc3 = wmma_f16(a, ld_ab(brow + 3 * bstride + k), acc3);
  }

  U8f u[4];
  u[0].v = acc0; u[1].v = acc1; u[2].v = acc2; u[3].v = acc3;
  #pragma unroll
  for (int j = 0; j < 4; ++j) {
    #pragma unroll
    for (int r = 0; r < 8; ++r) {
      const int m = mt * 16 + r + 8 * hf;   // C layout: VGPR r -> row r+8*half
      const int n = nt * 64 + j * 16 + row; // lane -> column
      if (Cf) {
        Cf[(size_t)m * N + n] = u[j].e[r];
      } else if (vtrans) {
        // m -> (b, s) ; n -> (h, dk) ; Vt[((b*16+h)*64+dk)*2048 + s]
        const int bb = m >> 11, ss = m & 2047;
        const int hh = n >> 6,  dk = n & 63;
        Ch[(((size_t)bb * 16 + hh) * 64 + dk) * 2048 + ss] = (_Float16)u[j].e[r];
      } else {
        Ch[(size_t)m * N + n] = (_Float16)u[j].e[r];
      }
    }
  }
}

// ---------------------------------------------------------------------------
// RoPE on Q and K in place (f16 storage, f32 math). Interleaved pairs.
// ---------------------------------------------------------------------------
__global__ void __launch_bounds__(256) rope_f16(_Float16* __restrict__ Q,
                                                _Float16* __restrict__ Kb,
                                                const int* __restrict__ pos,
                                                int total) {
  int idx = blockIdx.x * blockDim.x + threadIdx.x;
  if (idx >= total) return;
  const int i = idx & 31;
  const int h = (idx >> 5) & 15;
  const int s = (idx >> 9) & 2047;
  const int b = idx >> 20;
  const size_t off = ((size_t)b * 2048 + s) * 1024 + h * 64 + 2 * i;

  const float p = (float)pos[s];
  const float inv_freq = __powf(10000.0f, -(float)i * (1.0f / 32.0f));
  const float ang = p * inv_freq;
  float sn, c;
  __sincosf(ang, &sn, &c);

  const float q1 = (float)Q[off], q2 = (float)Q[off + 1];
  Q[off]     = (_Float16)(q1 * c - q2 * sn);
  Q[off + 1] = (_Float16)(q1 * sn + q2 * c);
  const float k1 = (float)Kb[off], k2 = (float)Kb[off + 1];
  Kb[off]     = (_Float16)(k1 * c - k2 * sn);
  Kb[off + 1] = (_Float16)(k1 * sn + k2 * c);
}

// ---------------------------------------------------------------------------
// Flash attention, causal. One wave per (b, h, 16-query tile).
// Key blocks of 32 keys, async-staged into LDS (double-buffered, one block
// ahead): K block 32x64 f16, Vt block 64x32 f16, 8 async b128 issues each.
// scores = 4 WMMAs, online softmax, P transposed C->A layout via LDS,
// P@V = 4 WMMAs. K layout [b][s][h*64+dk]; Vt layout [b][h][dk][s].
// ---------------------------------------------------------------------------
__global__ void __launch_bounds__(32) flash_attn_wmma(
    const _Float16* __restrict__ Q, const _Float16* __restrict__ Km,
    const _Float16* __restrict__ Vt, _Float16* __restrict__ O, int S) {
  constexpr int D = 1024;
  __shared__ _Float16 klds[2][32 * 64];  // [buf][key][dim]
  __shared__ _Float16 vlds[2][64 * 32];  // [buf][dim][key]
  __shared__ _Float16 plds[16 * 32];

  const int lane = threadIdx.x & 31;
  const int qt = blockIdx.x, h = blockIdx.y, b = blockIdx.z;
  const int row  = lane & 15;
  const int hf   = lane >> 4;
  const int koff = hf << 3;
  const int q0 = qt * 16;
  const size_t baseBH = (size_t)b * S * D + (size_t)h * 64;  // Q/K base
  const size_t baseVT = ((size_t)b * 16 + h) * 64 * (size_t)S;

  // Q tile in A layout, kept live for the whole key loop
  const _Float16* qrow = Q + baseBH + (size_t)(q0 + row) * D;
  const v16h qa0 = ld_ab(qrow + koff);       // dims 0..31
  const v16h qa1 = ld_ab(qrow + 32 + koff);  // dims 32..63

  // async stage of one 32-key block into LDS buffer `buf`
  auto stage_block = [&](int buf, int j0) {
    {  // K: 32 rows x 128B; 32 lanes x 16B = 4 rows/issue, 8 issues
      const int ro = lane >> 3;        // 0..3
      const int co = (lane & 7) * 8;   // half offset within row
      #pragma unroll
      for (int i = 0; i < 8; ++i) {
        const int key = 4 * i + ro;
        async_copy_b128((unsigned)(uintptr_t)&klds[buf][key * 64 + co],
                        Km + baseBH + (size_t)(j0 + key) * D + co);
      }
    }
    {  // Vt: 64 rows x 64B; 32 lanes x 16B = 8 rows/issue, 8 issues
      const int ro = lane >> 2;        // 0..7
      const int co = (lane & 3) * 8;
      #pragma unroll
      for (int i = 0; i < 8; ++i) {
        const int dim = 8 * i + ro;
        async_copy_b128((unsigned)(uintptr_t)&vlds[buf][dim * 32 + co],
                        Vt + baseVT + (size_t)dim * S + j0 + co);
      }
    }
  };

  U8f o[4];
  #pragma unroll
  for (int j = 0; j < 4; ++j) o[j].v = v8f{};
  float mrow[8], lrow[8];
  #pragma unroll
  for (int r = 0; r < 8; ++r) { mrow[r] = -3.0e38f; lrow[r] = 0.0f; }
  const float scale = 0.125f;  // 1/sqrt(DK=64)

  stage_block(0, 0);  // prologue prefetch

  for (int j0 = 0; j0 <= q0 + 15; j0 += 32) {
    const int buf = (j0 >> 5) & 1;
    const bool has_next = (j0 + 32) <= (q0 + 15);
    if (has_next) stage_block(buf ^ 1, j0 + 32);  // prefetch next block
    if (has_next) wait_async_16();                 // drain current block only
    else          wait_async_0();

    // scores: S(16x32) = Q(16x64) x K^T, two 16-key n-tiles from LDS
    const _Float16* kb = &klds[buf][0];
    U8f s0, s1;
    s0.v = v8f{}; s1.v = v8f{};
    s0.v = wmma_f16(qa0, ld_ab(kb + row * 64 + koff),                s0.v);
    s0.v = wmma_f16(qa1, ld_ab(kb + row * 64 + 32 + koff),           s0.v);
    s1.v = wmma_f16(qa0, ld_ab(kb + (16 + row) * 64 + koff),         s1.v);
    s1.v = wmma_f16(qa1, ld_ab(kb + (16 + row) * 64 + 32 + koff),    s1.v);

    #pragma unroll
    for (int r = 0; r < 8; ++r) {
      const int q = q0 + r + 8 * hf;  // query row for this C-VGPR
      float x0 = s0.e[r] * scale;
      float x1 = s1.e[r] * scale;
      if (j0 + row > q)      x0 = -3.0e38f;  // causal mask
      if (j0 + 16 + row > q) x1 = -3.0e38f;

      // row max over the 16 lanes of this half (xor<=8 stays in half)
      float mx = fmaxf(x0, x1);
      mx = fmaxf(mx, __shfl_xor(mx, 1, 32));
      mx = fmaxf(mx, __shfl_xor(mx, 2, 32));
      mx = fmaxf(mx, __shfl_xor(mx, 4, 32));
      mx = fmaxf(mx, __shfl_xor(mx, 8, 32));
      const float mnew  = fmaxf(mrow[r], mx);
      const float alpha = __expf(mrow[r] - mnew);
      const float p0 = __expf(x0 - mnew);
      const float p1 = __expf(x1 - mnew);
      float sum = p0 + p1;
      sum += __shfl_xor(sum, 1, 32);
      sum += __shfl_xor(sum, 2, 32);
      sum += __shfl_xor(sum, 4, 32);
      sum += __shfl_xor(sum, 8, 32);
      lrow[r] = lrow[r] * alpha + sum;
      mrow[r] = mnew;
      s0.e[r] = p0; s1.e[r] = p1;
      #pragma unroll
      for (int j = 0; j < 4; ++j) o[j].e[r] *= alpha;  // rescale accumulator
    }

    // transpose P from C layout to A layout through LDS
    #pragma unroll
    for (int r = 0; r < 8; ++r) {
      const int m = r + 8 * hf;
      plds[m * 32 + row]      = (_Float16)s0.e[r];
      plds[m * 32 + 16 + row] = (_Float16)s1.e[r];
    }
    __syncthreads();
    const v16h pa = ld_ab(&plds[row * 32 + koff]);
    __syncthreads();

    // O(16x64) += P(16x32) x V(32x64); B operand straight from LDS Vt tile
    #pragma unroll
    for (int dt = 0; dt < 4; ++dt) {
      const v16h vb = ld_ab(&vlds[buf][(dt * 16 + row) * 32 + koff]);
      o[dt].v = wmma_f16(pa, vb, o[dt].v);
    }
  }

  // normalize and store O tile (f16), layout [b][s][h*64+dk]
  #pragma unroll
  for (int r = 0; r < 8; ++r) {
    const float inv = 1.0f / lrow[r];
    const int m = q0 + r + 8 * hf;
    #pragma unroll
    for (int dt = 0; dt < 4; ++dt)
      O[baseBH + (size_t)m * D + dt * 16 + row] = (_Float16)(o[dt].e[r] * inv);
  }
}

// ---------------------------------------------------------------------------
// Host launcher
// ---------------------------------------------------------------------------
extern "C" void kernel_launch(void* const* d_in, const int* in_sizes, int n_in,
                              void* d_out, int out_size, void* d_ws, size_t ws_size,
                              hipStream_t stream) {
  (void)in_sizes; (void)n_in; (void)out_size; (void)ws_size;

  const float* x  = (const float*)d_in[0];
  const float* wq = (const float*)d_in[1];
  const float* wk = (const float*)d_in[2];
  const float* wv = (const float*)d_in[3];
  const float* wo = (const float*)d_in[4];
  const int* pos  = (const int*)d_in[5];

  constexpr int B = 4, S = 2048, D = 1024, H = 16;
  constexpr size_t NX = (size_t)B * S * D;  // 8,388,608 activations
  constexpr size_t NW = (size_t)D * D;      // 1,048,576 weight elems

  // workspace layout (all f16): xh | wq | wk | wv | wo | Q | K | Vt
  // O reuses xh region (xh dead after the three projections).
  char* ws = (char*)d_ws;
  _Float16* xh  = (_Float16*)(ws);
  _Float16* wqh = (_Float16*)(ws + 2 * NX);
  _Float16* wkh = (_Float16*)(ws + 2 * NX + 2 * NW);
  _Float16* wvh = (_Float16*)(ws + 2 * NX + 4 * NW);
  _Float16* woh = (_Float16*)(ws + 2 * NX + 6 * NW);
  _Float16* Qb  = (_Float16*)(ws + 2 * NX + 8 * NW);
  _Float16* Kb  = (_Float16*)(ws + 4 * NX + 8 * NW);
  _Float16* Vtb = (_Float16*)(ws + 6 * NX + 8 * NW);
  _Float16* Ob  = (_Float16*)(ws);  // alias of xh

  // 1) precision conversion
  f32_to_f16<<<(int)((NX + 255) / 256), 256, 0, stream>>>(x,  xh,  (int)NX);
  f32_to_f16<<<(int)((NW + 255) / 256), 256, 0, stream>>>(wq, wqh, (int)NW);
  f32_to_f16<<<(int)((NW + 255) / 256), 256, 0, stream>>>(wk, wkh, (int)NW);
  f32_to_f16<<<(int)((NW + 255) / 256), 256, 0, stream>>>(wv, wvh, (int)NW);
  f32_to_f16<<<(int)((NW + 255) / 256), 256, 0, stream>>>(wo, woh, (int)NW);

  // 2) Q/K/V projections: [8192,1024] @ [1024,1024]^T  (V stored transposed)
  const int waves  = ((B * S) / 16) * (D / 64);  // 8192 waves
  const int blocks = waves / 8;                  // 256 thr = 8 waves / block
  gemm_wmma_f16<<<blocks, 256, 0, stream>>>(xh, wqh, nullptr, Qb,  0, B * S, D, D);
  gemm_wmma_f16<<<blocks, 256, 0, stream>>>(xh, wkh, nullptr, Kb,  0, B * S, D, D);
  gemm_wmma_f16<<<blocks, 256, 0, stream>>>(xh, wvh, nullptr, Vtb, 1, B * S, D, D);

  // 3) RoPE on Q, K
  const int nr = B * S * H * 32;
  rope_f16<<<nr / 256, 256, 0, stream>>>(Qb, Kb, pos, nr);

  // 4) causal flash attention (async-LDS staged, double-buffered)
  flash_attn_wmma<<<dim3(S / 16, H, B), 32, 0, stream>>>(Qb, Kb, Vtb, Ob, S);

  // 5) output projection, f32 result
  gemm_wmma_f16<<<blocks, 256, 0, stream>>>(Ob, woh, (float*)d_out, nullptr, 0,
                                            B * S, D, D);
}
